// ConstrainedSWE_49538152792821
// MI455X (gfx1250) — matrix operand order repaired and checked by reading, not compiled, not run on hardware
//
#include <hip/hip_runtime.h>
#include <stdint.h>

typedef __attribute__((ext_vector_type(16))) _Float16 v16h;
typedef __attribute__((ext_vector_type(8)))  _Float16 h8;
typedef __attribute__((ext_vector_type(8)))  float    v8f;
typedef __attribute__((address_space(3)))    _Float16 lds_f16;

#define B_  4
#define N_  1024
#define D_  128
#define M_  1024
#define L_  16

// ---------------- workspace layout (bytes) ----------------
#define WS_W        0u            // 16*128 f32        = 8192
#define WS_XSL      8192u         // 4096*16 f32       = 262144
#define WS_XST      270336u       // 64*1024 f32       = 262144  (b,l,n) sorted
#define WS_RSL      532480u       // 1024*16 f32       = 65536
#define WS_RSORT    598016u       // 16*1024 f32       = 65536
#define WS_RIND     663552u       // 16*1024 i32       = 65536
#define WS_INVZX    729088u       // 64*1024 f32       = 262144
#define WS_PART     991232u       // 64*128 f32        = 32768
#define WS_COSTH    1024000u      // 4*1024*1024 f16   = 8388608
#define WS_SSR      9412608u      // 16*1024*1024 f16  = 33554432
// total ~41 MB

// ---------------- small helpers ----------------
__device__ inline float blk_reduce_sum(float v, float* red) {
  int lane = threadIdx.x & 31, w = threadIdx.x >> 5;
  #pragma unroll
  for (int o = 16; o > 0; o >>= 1) v += __shfl_xor(v, o, 32);
  if (lane == 0) red[w] = v;
  __syncthreads();
  int nw = blockDim.x >> 5;
  float s = (threadIdx.x < nw) ? red[threadIdx.x] : 0.f;
  if (w == 0) {
    #pragma unroll
    for (int o = 16; o > 0; o >>= 1) s += __shfl_xor(s, o, 32);
    if (lane == 0) red[0] = s;
  }
  __syncthreads();
  float r = red[0];
  __syncthreads();
  return r;
}

// ---------------- 1) weight_norm: W[l] = v[l]/||v[l]|| ----------------
__global__ void k_norm(const float* __restrict__ theta, float* __restrict__ W) {
  __shared__ float red[128];
  int l = blockIdx.x, t = threadIdx.x;        // 16 blocks x 128 threads
  float v = theta[l * D_ + t];
  red[t] = v * v;
  __syncthreads();
  for (int o = 64; o > 0; o >>= 1) { if (t < o) red[t] += red[t + o]; __syncthreads(); }
  W[l * D_ + t] = v * rsqrtf(red[0]);
}

// ---------------- 2) slices: out[row,l] = A[row,:] . W[l,:] ----------------
__global__ void k_slice(const float* __restrict__ A, const float* __restrict__ W,
                        float* __restrict__ out, int nrows) {
  int t = blockIdx.x * blockDim.x + threadIdx.x;
  if (t >= nrows * L_) return;
  int l = t & 15, row = t >> 4;
  const float* x = A + (size_t)row * D_;
  const float* w = W + l * D_;
  float s = 0.f;
  #pragma unroll 16
  for (int d = 0; d < D_; ++d) s = fmaf(x[d], w[d], s);
  out[t] = s;
}

// ---------------- 3) bitonic sort of Xsl along N per (b,l) ----------------
__global__ void k_sortx(const float* __restrict__ Xsl, float* __restrict__ XsT) {
  __shared__ float s[1024];
  int bx = blockIdx.x, t = threadIdx.x;       // 64 blocks x 512 threads
  int b = bx >> 4, l = bx & 15;
  s[t]       = Xsl[((b << 10) + t) * L_ + l];
  s[t + 512] = Xsl[((b << 10) + t + 512) * L_ + l];
  __syncthreads();
  for (int k = 2; k <= 1024; k <<= 1)
    for (int j = k >> 1; j > 0; j >>= 1) {
      int lo = t & (j - 1);
      int i0 = ((t ^ lo) << 1) | lo;
      int i1 = i0 | j;
      bool up = (i0 & k) == 0;
      float a = s[i0], c = s[i1];
      if ((a > c) == up) { s[i0] = c; s[i1] = a; }
      __syncthreads();
    }
  size_t base = (size_t)((b * L_ + l) << 10);
  XsT[base + t] = s[t];
  XsT[base + t + 512] = s[t + 512];
}

// ---------------- 4) bitonic argsort of Rsl per l ----------------
__global__ void k_argsortr(const float* __restrict__ Rsl, float* __restrict__ Rsort,
                           int* __restrict__ Rind) {
  __shared__ float sv[1024];
  __shared__ int   si[1024];
  int l = blockIdx.x, t = threadIdx.x;        // 16 blocks x 512 threads
  sv[t]       = Rsl[t * L_ + l];        si[t]       = t;
  sv[t + 512] = Rsl[(t + 512) * L_ + l]; si[t + 512] = t + 512;
  __syncthreads();
  for (int k = 2; k <= 1024; k <<= 1)
    for (int j = k >> 1; j > 0; j >>= 1) {
      int lo = t & (j - 1);
      int i0 = ((t ^ lo) << 1) | lo;
      int i1 = i0 | j;
      bool up = (i0 & k) == 0;
      float a = sv[i0], c = sv[i1];
      if ((a > c) == up) {
        sv[i0] = c; sv[i1] = a;
        int ia = si[i0]; si[i0] = si[i1]; si[i1] = ia;
      }
      __syncthreads();
    }
  Rsort[(l << 10) + t] = sv[t];        Rind[(l << 10) + t] = si[t];
  Rsort[(l << 10) + t + 512] = sv[t + 512]; Rind[(l << 10) + t + 512] = si[t + 512];
}

// ---------------- 5) embedding: out[b,l,m] = Rsl[m,l] - XsT[b,l,Rind[l,m]] ----------------
__global__ void k_emb(const float* __restrict__ Rsl, const int* __restrict__ Rind,
                      const float* __restrict__ XsT, float* __restrict__ out) {
  int t = blockIdx.x * 256 + threadIdx.x;     // 65536
  int b = t >> 14, l = (t >> 10) & 15, m = t & 1023;
  int idx = Rind[(l << 10) + m];
  out[t] = Rsl[m * L_ + l] - XsT[(size_t)(((b * L_ + l) << 10) + idx)];
}

// ---------------- 6) cost[b,n,m] = ||x_n - r_m|| -> f16 ----------------
__global__ void k_cost(const float* __restrict__ X, const float* __restrict__ R,
                       _Float16* __restrict__ costH) {
  __shared__ float Xt[16][128];
  __shared__ float Rt[16][128];
  int b = blockIdx.z, n0 = blockIdx.y << 4, m0 = blockIdx.x << 4;
  int tid = threadIdx.x;                      // 256
  for (int k = tid; k < 2048; k += 256) {
    int r = k >> 7, c = k & 127;
    Xt[r][c] = X[((size_t)((b << 10) + n0 + r)) * D_ + c];
    Rt[r][c] = R[(size_t)(m0 + r) * D_ + c];
  }
  __syncthreads();
  int nl = tid >> 4, ml = tid & 15;
  float d2 = 0.f;
  #pragma unroll 16
  for (int c = 0; c < 128; ++c) { float df = Xt[nl][c] - Rt[ml][c]; d2 = fmaf(df, df, d2); }
  float cst = sqrtf(fmaxf(d2, 1e-12f));
  costH[((size_t)((b << 10) + n0 + nl)) * 1024 + m0 + ml] = (_Float16)cst;
}

// ---------------- 7) materialize ss_r rows (f16) ----------------
__global__ void k_ssr(const float* __restrict__ Rsl, const float* __restrict__ Rsort,
                      _Float16* __restrict__ ssr) {
  __shared__ float e[1024];
  __shared__ float red[8];
  int bx = blockIdx.x, tid = threadIdx.x;     // 16384 blocks x 256
  int l = bx >> 10, i = bx & 1023;
  float c = Rsort[(l << 10) + 1023 - i];      // descending-sorted ref value
  float ps = 0.f;
  for (int m = tid; m < 1024; m += 256) {
    float df = Rsl[m * L_ + l] - c;
    float v = __expf(-df * df);
    e[m] = v; ps += v;
  }
  float tot = blk_reduce_sum(ps, red);
  float inv = 1.0f / tot;
  _Float16* dst = ssr + ((size_t)l << 20) + ((size_t)i << 10);
  for (int m = tid; m < 1024; m += 256) dst[m] = (_Float16)(e[m] * inv);
}

// ---------------- 8) invZ for ss_x rows ----------------
__global__ void k_invzx(const float* __restrict__ XsT, float* __restrict__ invZx) {
  __shared__ float red[8];
  int bx = blockIdx.x, tid = threadIdx.x;     // 4096 blocks x 256
  int d = bx >> 10, i = bx & 1023;
  const float* xs = XsT + ((size_t)d << 10);
  float c = xs[1023 - i];
  float ps = 0.f;
  for (int n = tid; n < 1024; n += 256) { float df = xs[n] - c; ps += __expf(-df * df); }
  float tot = blk_reduce_sum(ps, red);
  if (tid == 0) invZx[(d << 10) + i] = 1.0f / tot;
}

// ---------------- 9) fused WMMA: Ycr = cost @ ssr^T, dot with on-the-fly ss_x ----------------
// grid: x = 128 tiles (8 n-tiles x 16 i-tiles), y = 64 (d = b*16+l); block = 256 (8 waves)
// CDNA5 path: async DMA global->LDS (ASYNCcnt), double-buffered, one barrier/iter.
__global__ void __launch_bounds__(256)
k_plan(const _Float16* __restrict__ costH, const _Float16* __restrict__ ssr,
       const float* __restrict__ XsT, const float* __restrict__ invZx,
       float* __restrict__ partials) {
  __shared__ _Float16 As[2][128 * 32];   // cost tiles  (n x m), ping-pong
  __shared__ _Float16 Bs[2][64 * 32];    // ssr tiles   (i x m), ping-pong
  __shared__ float xsn[128];
  __shared__ float xrv[64];
  __shared__ float izv[64];
  __shared__ float red[8];

  int tid = threadIdx.x;
  int lane = tid & 31, w = tid >> 5;
  int tx = blockIdx.x;
  int n0 = (tx & 7) << 7;             // 0..896
  int i0 = (tx >> 3) << 6;            // 0..960
  int d = blockIdx.y, b = d >> 4, l = d & 15;

  v8f acc[4] = {};
  int hi16 = lane >> 4;               // 0/1
  int mrow = lane & 15;

  const _Float16* costB = costH + ((size_t)(b << 10)) * 1024;
  const _Float16* ssrL  = ssr + ((size_t)l << 20);

  // per-thread staging addresses (32 B per thread per tile)
  int srow = tid >> 1, scol = (tid & 1) << 4;
  const _Float16* gA_base = costB + (size_t)(n0 + srow) * 1024 + scol;
  const _Float16* gB_base = ssrL  + (size_t)(i0 + srow) * 1024 + scol;
  uint32_t ldsA[2], ldsB[2];
  ldsA[0] = (uint32_t)(uintptr_t)(lds_f16*)(&As[0][srow * 32 + scol]);
  ldsA[1] = (uint32_t)(uintptr_t)(lds_f16*)(&As[1][srow * 32 + scol]);
  ldsB[0] = (uint32_t)(uintptr_t)(lds_f16*)(&Bs[0][srow * 32 + scol]);
  ldsB[1] = (uint32_t)(uintptr_t)(lds_f16*)(&Bs[1][srow * 32 + scol]);

  auto stage = [&](int buf, int m0) {
    const _Float16* gA = gA_base + m0;
    asm volatile("global_load_async_to_lds_b128 %0, %1, off"
                 :: "v"(ldsA[buf]), "v"(gA) : "memory");
    asm volatile("global_load_async_to_lds_b128 %0, %1, off offset:16"
                 :: "v"(ldsA[buf]), "v"(gA) : "memory");
    if (tid < 128) {
      const _Float16* gB = gB_base + m0;
      asm volatile("global_load_async_to_lds_b128 %0, %1, off"
                   :: "v"(ldsB[buf]), "v"(gB) : "memory");
      asm volatile("global_load_async_to_lds_b128 %0, %1, off offset:16"
                   :: "v"(ldsB[buf]), "v"(gB) : "memory");
    }
  };

  // prologue: stage tile 0
  stage(0, 0);
  asm volatile("s_wait_asynccnt 0x0" ::: "memory");
  __syncthreads();

  for (int kk = 0; kk < 32; ++kk) {
    int cur = kk & 1;
    if (kk < 31) stage(cur ^ 1, (kk + 1) << 5);   // overlap DMA with WMMA

    // A fragment: M = 16*w + mrow, K = 16*hi16 + s (contiguous, matches B)
    const _Float16* ap = &As[cur][(16 * w + mrow) * 32];
    v16h a = *(const v16h*)(ap + 16 * hi16);

    #pragma unroll
    for (int tt = 0; tt < 4; ++tt) {
      const _Float16* bp = &Bs[cur][(16 * tt + mrow) * 32 + 16 * hi16];
      v16h bb = *(const v16h*)bp;
      acc[tt] = __builtin_amdgcn_wmma_f32_16x16x32_f16(
          false, a, false, bb, (short)0, acc[tt], false, false);
    }

    asm volatile("s_wait_asynccnt 0x0" ::: "memory");  // own DMA for next buf done
    __syncthreads();                                    // everyone done reading cur
  }

  // epilogue: multiply by ss_x[d, i, n] = exp(-(xs[n]-xs[1023-i])^2) * invZ[i], reduce
  if (tid < 128) xsn[tid] = XsT[((size_t)d << 10) + n0 + tid];
  if (tid < 64)  xrv[tid] = XsT[((size_t)d << 10) + 1023 - (i0 + tid)];
  if (tid >= 64 && tid < 128) { int i = tid - 64; izv[i] = invZx[(d << 10) + i0 + i]; }
  __syncthreads();

  float ps = 0.f;
  #pragma unroll
  for (int tt = 0; tt < 4; ++tt) {
    int ii = 16 * tt + mrow;
    float xr = xrv[ii], iz = izv[ii];
    #pragma unroll
    for (int r = 0; r < 8; ++r) {
      int nn = 16 * w + r + 8 * hi16;
      float df = xsn[nn] - xr;
      ps += acc[tt][r] * __expf(-df * df) * iz;
    }
  }
  float tot = blk_reduce_sum(ps, red);
  if (tid == 0) partials[(d << 7) + tx] = tot;
}

// ---------------- 10) finalize: per-slice distances ----------------
__global__ void k_final(const float* __restrict__ partials, float* __restrict__ out) {
  __shared__ float dd[64];
  int t = threadIdx.x;                        // 64
  float s = 0.f;
  for (int k = 0; k < 128; ++k) s += partials[(t << 7) + k];
  dd[t] = s;
  __syncthreads();
  if (t < 16)
    out[t] = (dd[t] + dd[16 + t] + dd[32 + t] + dd[48 + t]) * (1.0f / (1024.0f * 4.0f));
}

// ---------------- launcher ----------------
extern "C" void kernel_launch(void* const* d_in, const int* in_sizes, int n_in,
                              void* d_out, int out_size, void* d_ws, size_t ws_size,
                              hipStream_t stream) {
  const float* X     = (const float*)d_in[0];   // (4,1024,128)
  const float* theta = (const float*)d_in[1];   // (16,128)
  const float* ref   = (const float*)d_in[2];   // (1024,128)
  float* out = (float*)d_out;                   // 65536 emb + 16 dists

  char* ws = (char*)d_ws;
  float*     W     = (float*)(ws + WS_W);
  float*     Xsl   = (float*)(ws + WS_XSL);
  float*     XsT   = (float*)(ws + WS_XST);
  float*     Rsl   = (float*)(ws + WS_RSL);
  float*     Rsort = (float*)(ws + WS_RSORT);
  int*       Rind  = (int*)  (ws + WS_RIND);
  float*     invZx = (float*)(ws + WS_INVZX);
  float*     part  = (float*)(ws + WS_PART);
  _Float16*  costH = (_Float16*)(ws + WS_COSTH);
  _Float16*  ssr   = (_Float16*)(ws + WS_SSR);

  k_norm<<<16, 128, 0, stream>>>(theta, W);
  k_slice<<<(B_ * N_ * L_ + 255) / 256, 256, 0, stream>>>(X, W, Xsl, B_ * N_);
  k_slice<<<(M_ * L_ + 255) / 256, 256, 0, stream>>>(ref, W, Rsl, M_);
  k_sortx<<<B_ * L_, 512, 0, stream>>>(Xsl, XsT);
  k_argsortr<<<L_, 512, 0, stream>>>(Rsl, Rsort, Rind);
  k_emb<<<(B_ * L_ * M_) / 256, 256, 0, stream>>>(Rsl, Rind, XsT, out);
  k_cost<<<dim3(64, 64, B_), 256, 0, stream>>>(X, ref, costH);
  k_ssr<<<L_ * M_, 256, 0, stream>>>(Rsl, Rsort, ssr);
  k_invzx<<<B_ * L_ * N_ / 16, 256, 0, stream>>>(XsT, invZx);   // 4096 blocks
  k_plan<<<dim3(128, B_ * L_), 256, 0, stream>>>(costH, ssr, XsT, invZx, part);
  k_final<<<1, 64, 0, stream>>>(part, out + B_ * L_ * M_);
}
